// MultiHeadAttention_41583873360693
// MI455X (gfx1250) — compile-verified, matrix operand order checked by reading, not run on hardware
//
#include <hip/hip_runtime.h>
#include <hip/hip_bf16.h>

typedef __attribute__((ext_vector_type(16))) _Float16 v16h;
typedef __attribute__((ext_vector_type(8)))  _Float16 v8h;
typedef __attribute__((ext_vector_type(8)))  float    v8f;
typedef __attribute__((ext_vector_type(4)))  float    v4f;

#define D_MODEL 256
#define SEQ     2048
#define HEADS   8
#define DEPTH   32
#define BATCH   4
#define ROWS    (BATCH * SEQ)            // 8192
#define NEG_BIG (-1000000000.0f)
#define SCALE   0.17677669529663687f     // 1/sqrt(32)
#define SP      2052                     // padded LDS row stride (floats); %4==0 (16B ok), %64==4 -> bank spread

// ---------------------------------------------------------------------------
// WMMA helper (CDNA5 V_WMMA_F32_16X16X32_F16, wave32)
// ---------------------------------------------------------------------------
__device__ __forceinline__ v8f wmma16x16x32(v16h a, v16h b, v8f c) {
  // (neg_a, A, neg_b, B, c_mod, C, reuse_a, reuse_b)
  return __builtin_amdgcn_wmma_f32_16x16x32_f16(false, a, false, b, (short)0, c, false, false);
}

// A/B fragment from a row-major f16 matrix. Per ISA 7.12.2 (16-bit A 16x32):
// lanes 0-15 hold row=lane, K chunks {k0+0..7, k0+16..23};
// lanes 16-31 hold row=lane-16, K chunks {k0+8..15, k0+24..31}.
__device__ __forceinline__ v16h frag_f16(const _Float16* base, int row, int stride,
                                         int k0, int laneHi) {
  const _Float16* p = base + (size_t)row * stride + k0 + (laneHi ? 8 : 0);
  v8h lo = *(const v8h*)(p);
  v8h hi = *(const v8h*)(p + 16);
  v16h r;
#pragma unroll
  for (int i = 0; i < 8; ++i) { r[i] = lo[i]; r[8 + i] = hi[i]; }
  return r;
}

// Same fragment but converting from an f32 row-major source (global).
__device__ __forceinline__ v16h frag_f32cvt(const float* base, int row, int stride,
                                            int k0, int laneHi) {
  const float* p = base + (size_t)row * stride + k0 + (laneHi ? 8 : 0);
  v16h r;
#pragma unroll
  for (int i = 0; i < 8; ++i) { r[i] = (_Float16)p[i]; r[8 + i] = (_Float16)p[i + 16]; }
  return r;
}

// ---------------------------------------------------------------------------
// Kernel 1: transpose+convert the four weight matrices to f16 [n][k]
// ---------------------------------------------------------------------------
__global__ void mha_wprep(const float* __restrict__ Wq, const float* __restrict__ Wk,
                          const float* __restrict__ Wv, const float* __restrict__ Wo,
                          _Float16* __restrict__ WT) {
  int idx = blockIdx.x * 256 + threadIdx.x;      // 4 * 65536 total
  int w = idx >> 16;
  int e = idx & 65535;
  int kk = e >> 8;          // source row (k)
  int n  = e & 255;         // source col (n)
  const float* src = (w == 0) ? Wq : (w == 1) ? Wk : (w == 2) ? Wv : Wo;
  WT[(size_t)w * 65536 + (size_t)n * 256 + kk] = (_Float16)src[(size_t)kk * 256 + n];
}

// ---------------------------------------------------------------------------
// Kernel 2: fused Q/K/V projections. One 16x16 output tile per wave.
//   Qh, Kh : [bh][s][DEPTH]   f16 (row-major per head)
//   Vt     : [bh][DEPTH][s]   f16 (transposed so attn@V B-frags are contiguous)
// ---------------------------------------------------------------------------
__global__ void mha_proj(const float* __restrict__ q, const float* __restrict__ k,
                         const float* __restrict__ v,
                         const _Float16* __restrict__ WqT, const _Float16* __restrict__ WkT,
                         const _Float16* __restrict__ WvT,
                         _Float16* __restrict__ Qh, _Float16* __restrict__ Kh,
                         _Float16* __restrict__ Vt) {
  const int MT = ROWS / 16;          // 512
  const int NT = D_MODEL / 16;       // 16
  int wid  = blockIdx.x * 8 + (threadIdx.x >> 5);
  int proj = wid / (MT * NT);
  int rem  = wid % (MT * NT);
  int mt = rem / NT, nt = rem % NT;
  const float*    X = (proj == 0) ? q : (proj == 1) ? k : v;
  const _Float16* W = (proj == 0) ? WqT : (proj == 1) ? WkT : WvT;

  int lane = threadIdx.x & 31, lrow = lane & 15, lhi = lane >> 4;
  v8f c = 0.f;
#pragma unroll
  for (int kk = 0; kk < D_MODEL; kk += 32) {
    v16h a = frag_f32cvt(X, mt * 16 + lrow, D_MODEL, kk, lhi);
    v16h b = frag_f16(W, nt * 16 + lrow, D_MODEL, kk, lhi);   // W^T row n = column n
    c = wmma16x16x32(a, b, c);
  }

  int col = nt * 16 + lrow;          // 0..255 model dim
  int h = col >> 5, d = col & 31;
  if (proj < 2) {
    _Float16* dst = (proj == 0) ? Qh : Kh;
#pragma unroll
    for (int r = 0; r < 8; ++r) {
      int gm = mt * 16 + r + 8 * lhi;          // global row in [0,8192)
      int b = gm >> 11, s = gm & 2047;
      dst[(((size_t)(b * HEADS + h) * SEQ) + s) * DEPTH + d] = (_Float16)c[r];
    }
  } else {
#pragma unroll
    for (int r = 0; r < 8; ++r) {
      int gm = mt * 16 + r + 8 * lhi;
      int b = gm >> 11, s = gm & 2047;
      Vt[((size_t)(b * HEADS + h) * DEPTH + d) * SEQ + s] = (_Float16)c[r];
    }
  }
}

// ---------------------------------------------------------------------------
// Kernel 3: fused attention. One WG (256 thr / 8 waves) per (bh, 16-query block).
//   - logits = Q Kt * scale + mask  (WMMA, K=32=depth, one wmma per 16x16 tile)
//   - softmax over 2048 keys in LDS (float4 throughout)
//   - write normalized attn to HBM exactly once (256B coalesced float4 bursts)
//   - ctx = attn @ V via WMMA with cross-wave LDS reduction -> concat f16
// ---------------------------------------------------------------------------
__global__ void mha_attn(const _Float16* __restrict__ Qh, const _Float16* __restrict__ Kh,
                         const _Float16* __restrict__ Vt, const float* __restrict__ mask,
                         float* __restrict__ attn, _Float16* __restrict__ concat) {
  extern __shared__ float lds[];
  float* P   = lds;                        // [16][SP] logits / probabilities
  float* CT  = lds + 16 * SP;              // [8][512] per-wave ctx partials
  float* RED = CT + 8 * 512;               // [16][16] reduction scratch

  int bh = blockIdx.x >> 7;                // 0..31
  int qt = blockIdx.x & 127;               // query block
  int b  = bh >> 3;
  int h  = bh & 7;
  int wave = threadIdx.x >> 5, lane = threadIdx.x & 31;
  int lrow = lane & 15, lhi = lane >> 4;

  const _Float16* Qb = Qh + (size_t)bh * SEQ * DEPTH;
  const _Float16* Kb = Kh + (size_t)bh * SEQ * DEPTH;
  const _Float16* Vb = Vt + (size_t)bh * DEPTH * SEQ;

  // The Q fragment is shared by every key tile: load once.
  v16h aq = frag_f16(Qb, qt * 16 + lrow, DEPTH, 0, lhi);

  // ---- logits: 128 key tiles split across 8 waves ----
  for (int kt = wave; kt < 128; kt += 8) {
    if (kt + 8 < 128)                      // gfx1250 global_prefetch_b8 of next K tile
      __builtin_prefetch(Kb + (size_t)(kt + 8) * 16 * DEPTH + lane * 16, 0, 3);
    v16h bk = frag_f16(Kb, kt * 16 + lrow, DEPTH, 0, lhi);
    v8f c = 0.f;
    c = wmma16x16x32(aq, bk, c);
    int kcol = kt * 16 + lrow;               // D frag: N = lrow
    float mterm = (1.0f - mask[b * SEQ + kcol]) * NEG_BIG;
#pragma unroll
    for (int r = 0; r < 8; ++r)              // D frag: M = r + 8*lhi
      P[(r + 8 * lhi) * SP + kcol] = c[r] * SCALE + mterm;
  }
  __syncthreads();

  // ---- softmax: 16 threads per row, float4 lanes ----
  int row = threadIdx.x >> 4;                // 0..15
  int j   = threadIdx.x & 15;
  v4f* Prow4 = (v4f*)&P[row * SP];           // SP%4==0, 16B aligned

  float m = -3.4e38f;
  for (int i = j; i < SEQ / 4; i += 16) {
    v4f x = Prow4[i];
    m = fmaxf(m, fmaxf(fmaxf(x[0], x[1]), fmaxf(x[2], x[3])));
  }
  RED[row * 16 + j] = m;
  __syncthreads();
  m = RED[row * 16];
#pragma unroll
  for (int i = 1; i < 16; ++i) m = fmaxf(m, RED[row * 16 + i]);

  float sum = 0.f;
  for (int i = j; i < SEQ / 4; i += 16) {
    v4f x = Prow4[i];
    x[0] = __expf(x[0] - m); x[1] = __expf(x[1] - m);
    x[2] = __expf(x[2] - m); x[3] = __expf(x[3] - m);
    Prow4[i] = x;
    sum += (x[0] + x[1]) + (x[2] + x[3]);
  }
  __syncthreads();                            // everyone done reading RED (max)
  RED[row * 16 + j] = sum;
  __syncthreads();
  sum = 0.f;
#pragma unroll
  for (int i = 0; i < 16; ++i) sum += RED[row * 16 + i];
  float inv = 1.0f / sum;

  // normalize in LDS and emit attn: 16 consecutive lanes -> 256B contiguous stores
  v4f* arow4 = (v4f*)(attn + ((size_t)bh * SEQ + qt * 16 + row) * SEQ);
  for (int i = j; i < SEQ / 4; i += 16) {
    v4f x = Prow4[i];
    x *= inv;
    Prow4[i] = x;
    arow4[i] = x;                             // the single HBM write of attn
  }
  __syncthreads();                            // P finalized for all waves

  // ---- ctx = P(16x2048) @ V(2048x32): 64 K-tiles of 32 split across 8 waves ----
  v8f c0 = 0.f, c1 = 0.f;
  for (int kt = wave; kt < 64; kt += 8) {
    int k0 = kt * 32;
    if (kt + 8 < 64)
      __builtin_prefetch(Vb + (size_t)lrow * SEQ + (kt + 8) * 32, 0, 3);
    v16h a;                                   // A frag from LDS probabilities, cvt f16
    {
      const float* p = &P[lrow * SP + k0 + (lhi ? 8 : 0)];
      v4f x0 = *(const v4f*)(p);
      v4f x1 = *(const v4f*)(p + 4);
      v4f x2 = *(const v4f*)(p + 16);
      v4f x3 = *(const v4f*)(p + 20);
#pragma unroll
      for (int i = 0; i < 4; ++i) {
        a[i]      = (_Float16)x0[i];
        a[4 + i]  = (_Float16)x1[i];
        a[8 + i]  = (_Float16)x2[i];
        a[12 + i] = (_Float16)x3[i];
      }
    }
    v16h b0 = frag_f16(Vb, lrow,      SEQ, k0, lhi);   // d columns 0..15
    v16h b1 = frag_f16(Vb, 16 + lrow, SEQ, k0, lhi);   // d columns 16..31
    c0 = wmma16x16x32(a, b0, c0);
    c1 = wmma16x16x32(a, b1, c1);
  }
#pragma unroll
  for (int r = 0; r < 8; ++r) {
    int M = r + 8 * lhi;
    CT[wave * 512 + M * 32 + lrow]      = c0[r];
    CT[wave * 512 + M * 32 + 16 + lrow] = c1[r];
  }
  __syncthreads();

  // reduce the 8 per-wave partials and emit concat[b][s][h*32+d] as f16
  for (int e = threadIdx.x; e < 512; e += 256) {
    float s = 0.f;
#pragma unroll
    for (int w = 0; w < 8; ++w) s += CT[w * 512 + e];
    int M = e >> 5, col = e & 31;
    int srow = qt * 16 + M;
    concat[((size_t)(b * SEQ + srow)) * D_MODEL + h * DEPTH + col] = (_Float16)s;
  }
}

// ---------------------------------------------------------------------------
// Kernel 4: out = concat @ Wo. One 16x16 tile per wave.
// ---------------------------------------------------------------------------
__global__ void mha_outproj(const _Float16* __restrict__ concat,
                            const _Float16* __restrict__ WoT,
                            float* __restrict__ out) {
  int wid = blockIdx.x * 8 + (threadIdx.x >> 5);
  int mt = wid >> 4, nt = wid & 15;            // 512 x 16 tiles
  int lane = threadIdx.x & 31, lrow = lane & 15, lhi = lane >> 4;
  v8f c = 0.f;
#pragma unroll
  for (int kk = 0; kk < D_MODEL; kk += 32) {
    v16h a = frag_f16(concat, mt * 16 + lrow, D_MODEL, kk, lhi);
    v16h b = frag_f16(WoT,    nt * 16 + lrow, D_MODEL, kk, lhi);
    c = wmma16x16x32(a, b, c);
  }
#pragma unroll
  for (int r = 0; r < 8; ++r) {
    int m = mt * 16 + r + 8 * lhi;
    out[(size_t)m * D_MODEL + nt * 16 + lrow] = c[r];
  }
}

// ---------------------------------------------------------------------------
// Host launcher
// ---------------------------------------------------------------------------
extern "C" void kernel_launch(void* const* d_in, const int* in_sizes, int n_in,
                              void* d_out, int out_size, void* d_ws, size_t ws_size,
                              hipStream_t stream) {
  (void)in_sizes; (void)n_in; (void)out_size; (void)ws_size;
  const float* v    = (const float*)d_in[0];
  const float* k    = (const float*)d_in[1];
  const float* q    = (const float*)d_in[2];
  const float* mask = (const float*)d_in[3];
  const float* Wq   = (const float*)d_in[4];
  const float* Wk   = (const float*)d_in[5];
  const float* Wv   = (const float*)d_in[6];
  const float* Wo   = (const float*)d_in[7];

  // workspace layout (f16): WT[4][256][256] | Qh | Kh | Vt | concat  (~16.5 MB)
  char* ws = (char*)d_ws;
  _Float16* WT     = (_Float16*)ws;                       // 4 * 65536
  _Float16* Qh     = (_Float16*)(ws + 512 * 1024);        // each 2,097,152 f16 (4 MB)
  _Float16* Kh     = Qh + 2097152;
  _Float16* Vt     = Kh + 2097152;
  _Float16* concat = Vt + 2097152;

  float* out  = (float*)d_out;                            // 8192*256
  float* attn = out + (size_t)ROWS * D_MODEL;             // 4*8*2048*2048

  // 1) weights -> f16, transposed
  mha_wprep<<<dim3(1024), dim3(256), 0, stream>>>(Wq, Wk, Wv, Wo, WT);

  // 2) fused QKV projections: 3 * 512 * 16 tiles / 8 waves per block
  mha_proj<<<dim3(3072), dim3(256), 0, stream>>>(q, k, v, WT, WT + 65536, WT + 2 * 65536,
                                                 Qh, Kh, Vt);

  // 3) fused attention (~145 KB LDS: legal on CDNA5, 320 KB/WG)
  size_t shmem = (size_t)(16 * SP + 8 * 512 + 16 * 16) * sizeof(float);
  hipFuncSetAttribute((const void*)mha_attn,
                      hipFuncAttributeMaxDynamicSharedMemorySize, (int)shmem);
  mha_attn<<<dim3(4096), dim3(256), shmem, stream>>>(Qh, Kh, Vt, mask, attn, concat);

  // 4) output projection: 512*16 tiles / 8 waves per block
  mha_outproj<<<dim3(1024), dim3(256), 0, stream>>>(concat, WT + 3 * 65536, out);
}
